// BiLSTMDecoderModel_575525617779
// MI455X (gfx1250) — compile-verified, matrix operand order checked by reading, not run on hardware
//
#include <hip/hip_runtime.h>
#include <hip/hip_bf16.h>
#include <math.h>

// ---------------------------------------------------------------------------
// BiLSTM + GRU decoder for gfx1250 (MI455X), wave32 + WMMA.
//
// Algebraic reductions vs. the reference:
//  * backward LSTM scan collapses to ONE cell step on seq_emb[0] with h=c=0
//    (b_Whh is mathematically unused).
//  * forward-LSTM x@Wih.T precomputed for all 256 steps as one big WMMA GEMM,
//    stored as f16 in the C/D per-lane fragment layout (halves HBM traffic of
//    the dominant buffer and makes producer stores / consumer loads b128).
//  * decoder gi = ce@d_Wih.T is batch-invariant -> precomputed [6,3072].
// Recurrences run in persistent kernels with Whh slabs resident in LDS
// (128KB / 192KB per WGP) and a global monotonic barrier per step.
// All matrix operands are pre-packed into the CDNA5 WMMA per-lane fragment
// layout so hot-loop fragment loads are contiguous b128 loads.
// ---------------------------------------------------------------------------

typedef __attribute__((ext_vector_type(16))) _Float16 v16h;
typedef __attribute__((ext_vector_type(8)))  _Float16 v8h;
typedef __attribute__((ext_vector_type(8)))  float    v8f;

constexpr int Bn = 128, Sn = 256, En = 300, EP = 320, Hn = 512;
constexpr int G4 = 2048, H2 = 1024, G6 = 3072, NC = 6;
constexpr int KT_E  = EP / 32;   // 10
constexpr int KT_H  = Hn / 32;   // 16
constexpr int KT_H2 = H2 / 32;   // 32
constexpr int NROWS = Sn * Bn;   // 32768
constexpr int NWGR  = 16;        // persistent workgroups for LSTM recurrence
constexpr int NT_G4 = G4 / 16;   // 128 n-tiles of the 4H gate matrix

__device__ __forceinline__ float sigf(float x) { return 1.0f / (1.0f + expf(-x)); }

// CDNA5 16-bit A/B fragment element mapping (16x32 tile, wave32):
// lanes 0-15 hold rows, lanes 16-31 same rows with K shifted by 8.
__device__ __forceinline__ int frag_lane(int mn, int k) {
  return (mn & 15) + (((k >> 3) & 1) << 4);
}
__device__ __forceinline__ int frag_elem(int k) {
  int kg = (k >> 3) & 1;
  int b  = k - 8 * kg;
  return (b >= 16) ? (b - 8) : b;
}

// ---------------------------------------------------------------------------
// Workspace init: zero barrier + zero h0 (packed f16 zeros are bit-zero).
__global__ void k_init(int* bar, unsigned int* hb0, int nwords) {
  int tid = blockIdx.x * blockDim.x + threadIdx.x;
  if (tid == 0) bar[0] = 0;
  if (tid < nwords) hb0[tid] = 0u;
}

// ---------------------------------------------------------------------------
// Pack W^T (W row-major [N][Kreal], f32) into f16 B-fragment layout
// index = ((ntile*KTiles + ktile)*32 + lane)*16 + elem.  Zero-pads K.
__global__ void k_pack_b(const float* __restrict__ W, int N, int Kreal, int KTiles,
                         _Float16* __restrict__ out) {
  long tid   = (long)blockIdx.x * blockDim.x + threadIdx.x;
  int  Kpad  = KTiles * 32;
  long total = (long)N * Kpad;
  if (tid >= total) return;
  int n = (int)(tid / Kpad), k = (int)(tid % Kpad);
  float v = (k < Kreal) ? W[(long)n * Kreal + k] : 0.0f;
  int nt = n >> 4, kt = k >> 5, kk = k & 31;
  out[(((long)nt * KTiles + kt) * 32 + frag_lane(n, kk)) * 16 + frag_elem(kk)] = (_Float16)v;
}

// ---------------------------------------------------------------------------
// Gather embedding rows, tanh, convert to f16, scatter into packed A-fragment
// layout.  Row index = s*128 + b  (seq is [B,S] row-major).
__global__ void k_embed_pack(const int* __restrict__ seq, const float* __restrict__ embW,
                             _Float16* __restrict__ x16) {
  long tid = (long)blockIdx.x * blockDim.x + threadIdx.x;
  long total = (long)NROWS * EP;
  if (tid >= total) return;
  int row = (int)(tid / EP), col = (int)(tid % EP);
  int s = row >> 7, b = row & 127;
  float v = 0.0f;
  if (col < En) {
    int tok = seq[b * Sn + s];
    v = tanhf(embW[(long)tok * En + col]);
  }
  int mt = row >> 4, kt = col >> 5, kk = col & 31;
  x16[(((long)mt * KT_E + kt) * 32 + frag_lane(row, kk)) * 16 + frag_elem(kk)] = (_Float16)v;
}

// ---------------------------------------------------------------------------
// Generic packed WMMA GEMM: A(f16 frags) @ B(f16 frags) + bias.
// Grid: (M/128, N/128); 256 threads = 8 waves; wave = one 16-row M tile,
// each wave accumulates 8 N-tiles (64 VGPRs of f32 accumulators).
// packedOut==0: f32 row-major [M][Nstride].
// packedOut==1: f16 C/D-fragment layout ((gm*NTtot + gn)*32 + lane)*8 + r
//               (one contiguous b128 store per tile per lane).
__global__ void __launch_bounds__(256)
k_gemm_bias(const _Float16* __restrict__ Ap, const _Float16* __restrict__ Bp,
            const float* __restrict__ bias, float* __restrict__ outF,
            _Float16* __restrict__ outP, int KTiles, int Nstride, int packedOut) {
  int lane = threadIdx.x & 31, wv = threadIdx.x >> 5;
  int n16 = lane & 15, hi = lane >> 4;
  int gm = blockIdx.x * 8 + wv;
  int NTtot = gridDim.y * 8;
  v8f acc[8];
#pragma unroll
  for (int nt = 0; nt < 8; ++nt) { v8f z = {}; acc[nt] = z; }
  for (int kt = 0; kt < KTiles; ++kt) {
    v16h a = *(const v16h*)(Ap + (((long)gm * KTiles + kt) * 32 + lane) * 16);
#pragma unroll
    for (int nt = 0; nt < 8; ++nt) {
      int gn = blockIdx.y * 8 + nt;
      v16h b = *(const v16h*)(Bp + (((long)gn * KTiles + kt) * 32 + lane) * 16);
      acc[nt] = __builtin_amdgcn_wmma_f32_16x16x32_f16(false, a, false, b,
                                                       (short)0, acc[nt], false, false);
    }
  }
  if (packedOut) {
#pragma unroll
    for (int nt = 0; nt < 8; ++nt) {
      int gn = blockIdx.y * 8 + nt;
      float bv = bias[gn * 16 + n16];
      v8h o;
#pragma unroll
      for (int r = 0; r < 8; ++r) o[r] = (_Float16)(acc[nt][r] + bv);
      *(v8h*)(outP + (((long)gm * NTtot + gn) * 32 + lane) * 8) = o;
    }
  } else {
    int row0 = blockIdx.x * 128 + wv * 16 + hi * 8;
#pragma unroll
    for (int nt = 0; nt < 8; ++nt) {
      int gn  = blockIdx.y * 8 + nt;
      int col = gn * 16 + n16;
      float bv = bias[col];
#pragma unroll
      for (int r = 0; r < 8; ++r)
        outF[(long)(row0 + r) * Nstride + col] = acc[nt][r] + bv;
    }
  }
}

// ---------------------------------------------------------------------------
// Backward "scan" == one LSTM cell on XB (gates already = x@bWih.T + b_b), h=c=0.
// Writes h into hf32[:,512:1024] and packed hdec (ktiles 16..31).
__global__ void k_bwd_cell(const float* __restrict__ XB,
                           _Float16* __restrict__ hdec, float* __restrict__ hf32) {
  int tid = blockIdx.x * blockDim.x + threadIdx.x;
  if (tid >= Bn * Hn) return;
  int b = tid >> 9, j = tid & 511;
  const float* g = XB + (long)b * G4;
  float iv = g[j], gv = g[1024 + j], ov = g[1536 + j];
  float cn = sigf(iv) * tanhf(gv);       // sigmoid(f)*c_prev == 0
  float h  = sigf(ov) * tanhf(cn);
  hf32[(long)b * H2 + 512 + j] = h;
  int kd = 512 + j, kt = kd >> 5, kk = kd & 31;
  hdec[(((long)(b >> 4) * KT_H2 + kt) * 32 + frag_lane(b, kk)) * 16 + frag_elem(kk)] = (_Float16)h;
}

// ---------------------------------------------------------------------------
// Persistent forward-LSTM recurrence.  16 workgroups (one per WGP), each owns
// 32 h-columns: Whh^T slab (512 x 4*32 gate cols, f16 = 128KB) resident in LDS,
// c state (128x32 f32 = 16KB) in LDS.  h ping-pongs via L2 in packed A layout;
// per-step sync = monotonic global barrier with s_sleep backoff.
// XGp is f16 in C/D-fragment layout -> accumulator seed is one b128 load + cvt.
__global__ void __launch_bounds__(256, 1)
k_lstm_fwd(const _Float16* __restrict__ whhP, const _Float16* __restrict__ XGp,
           _Float16* __restrict__ hb0, _Float16* __restrict__ hb1,
           _Float16* __restrict__ hdec, float* __restrict__ hf32,
           int* __restrict__ bar) {
  extern __shared__ char smem[];
  _Float16* Wl  = (_Float16*)smem;                        // 8*16*512 halves = 128KB
  float*    cms = (float*)(smem + (size_t)8 * KT_H * 512 * 2);  // 128*32 f32 = 16KB
  const int tid = threadIdx.x, lane = tid & 31, wv = tid >> 5, w = blockIdx.x;
  const int n16 = lane & 15, hi = lane >> 4;

  // Preload weight slab: local nt = 2*gate + t  ->  global ntile 32*g + 2*w + t.
  for (int nt = 0; nt < 8; ++nt) {
    int g = nt >> 1, t = nt & 1, gn = 32 * g + 2 * w + t;
    const uint4* src = (const uint4*)(whhP + (long)gn * KT_H * 32 * 16);
    uint4*       dst = (uint4*)(Wl + (long)nt * KT_H * 32 * 16);
    for (int i = tid; i < KT_H * 32 * 16 / 8; i += 256) dst[i] = src[i];
  }
  for (int i = tid; i < 128 * 32; i += 256) cms[i] = 0.0f;
  __syncthreads();

  for (int s = 0; s < Sn; ++s) {
    const _Float16* hcur = (s & 1) ? hb1 : hb0;
    _Float16*       hnxt = (s & 1) ? hb0 : hb1;

    // Accumulators seeded with precomputed x@Wih.T + f_b gate tiles (packed f16).
    v8f acc[8];
#pragma unroll
    for (int nt = 0; nt < 8; ++nt) {
      int g = nt >> 1, t = nt & 1;
      int gn = 32 * g + 2 * w + t;                       // global gate n-tile
      long gm = (long)s * 8 + wv;                        // global m-tile
      v8h xv = *(const v8h*)(XGp + ((gm * NT_G4 + gn) * 32 + lane) * 8);
      v8f c;
#pragma unroll
      for (int r = 0; r < 8; ++r) c[r] = (float)xv[r];
      acc[nt] = c;
    }

    // gates += h @ Whh^T (slice): 8 Mtiles x 8 Ntiles x 16 Ktiles per WG.
#pragma unroll 4
    for (int kt = 0; kt < KT_H; ++kt) {
      v16h a = *(const v16h*)(hcur + (((long)wv * KT_H + kt) * 32 + lane) * 16);
#pragma unroll
      for (int nt = 0; nt < 8; ++nt) {
        v16h b = *(const v16h*)(Wl + (((long)nt * KT_H + kt) * 32 + lane) * 16);
        acc[nt] = __builtin_amdgcn_wmma_f32_16x16x32_f16(false, a, false, b,
                                                         (short)0, acc[nt], false, false);
      }
    }

    // LSTM cell elementwise; write new h (f16, packed A layout) for next step.
#pragma unroll
    for (int t = 0; t < 2; ++t) {
#pragma unroll
      for (int r = 0; r < 8; ++r) {
        int brow = wv * 16 + r + 8 * hi;
        int jl   = t * 16 + n16;
        float iv = acc[0 + t][r], fv = acc[2 + t][r];
        float gv = acc[4 + t][r], ov = acc[6 + t][r];
        float cp = cms[brow * 32 + jl];
        float cn = sigf(fv) * cp + sigf(iv) * tanhf(gv);
        float h  = sigf(ov) * tanhf(cn);
        cms[brow * 32 + jl] = cn;
        int lp = frag_lane(brow, jl), e = frag_elem(jl);
        hnxt[(((long)wv * KT_H + w) * 32 + lp) * 16 + e] = (_Float16)h;
        if (s == Sn - 1) {
          hdec[(((long)wv * KT_H2 + w) * 32 + lp) * 16 + e] = (_Float16)h;
          hf32[(long)brow * H2 + 32 * w + jl] = h;
        }
      }
    }

    // Cross-WG step barrier (monotonic counter, reset by k_init each launch).
    __threadfence();
    __syncthreads();
    if (tid == 0) {
      atomicAdd(bar, 1);
      int want = NWGR * (s + 1);
      volatile int* vb = bar;
      while (*vb < want) __builtin_amdgcn_s_sleep(4);
      __threadfence();
    }
    __syncthreads();
  }
}

// ---------------------------------------------------------------------------
// GI[k][col] = tanh(embed_class_W[classes[k]]) . d_Wih[col] + d_bih[col]
__global__ void k_gi(const int* __restrict__ classes, const float* __restrict__ ecW,
                     const float* __restrict__ dWih, const float* __restrict__ dbih,
                     float* __restrict__ GI) {
  int tid = blockIdx.x * blockDim.x + threadIdx.x;
  if (tid >= NC * G6) return;
  int k = tid / G6, col = tid % G6;
  const float* er = ecW + (long)classes[k] * Hn;
  const float* wr = dWih + (long)col * Hn;
  float a = dbih[col];
  for (int j = 0; j < Hn; ++j) a += tanhf(er[j]) * wr[j];
  GI[tid] = a;
}

// ---------------------------------------------------------------------------
// One GRU-decoder step.  32 WGs, each owns 32 h-columns (3 gate strips r/z/n).
// d_Whh^T slab (1024 x 96 f16 = 192KB) in LDS.  gi kept separate from gh so
// r multiplies only the hidden part of the n gate.
__global__ void __launch_bounds__(256, 1)
k_dec_step(const _Float16* __restrict__ dwhhP, const float* __restrict__ GIk,
           const float* __restrict__ dbhh,
           const _Float16* __restrict__ hdec_cur, const float* __restrict__ hf_cur,
           _Float16* __restrict__ hdec_nxt, float* __restrict__ hf_nxt) {
  extern __shared__ char smem[];
  _Float16* Wl = (_Float16*)smem;   // 6*32*512 halves = 192KB
  const int tid = threadIdx.x, lane = tid & 31, wv = tid >> 5, w = blockIdx.x;
  const int n16 = lane & 15, hi = lane >> 4;

  for (int nt = 0; nt < 6; ++nt) {
    int g = nt >> 1, t = nt & 1, gn = 64 * g + 2 * w + t;
    const uint4* src = (const uint4*)(dwhhP + (long)gn * KT_H2 * 32 * 16);
    uint4*       dst = (uint4*)(Wl + (long)nt * KT_H2 * 32 * 16);
    for (int i = tid; i < KT_H2 * 32 * 16 / 8; i += 256) dst[i] = src[i];
  }
  __syncthreads();

  v8f acc[6];
#pragma unroll
  for (int nt = 0; nt < 6; ++nt) {
    int g = nt >> 1, t = nt & 1;
    float bv = dbhh[1024 * g + 32 * w + t * 16 + n16];
    v8f c;
#pragma unroll
    for (int r = 0; r < 8; ++r) c[r] = bv;
    acc[nt] = c;
  }
#pragma unroll 4
  for (int kt = 0; kt < KT_H2; ++kt) {
    v16h a = *(const v16h*)(hdec_cur + (((long)wv * KT_H2 + kt) * 32 + lane) * 16);
#pragma unroll
    for (int nt = 0; nt < 6; ++nt) {
      v16h b = *(const v16h*)(Wl + (((long)nt * KT_H2 + kt) * 32 + lane) * 16);
      acc[nt] = __builtin_amdgcn_wmma_f32_16x16x32_f16(false, a, false, b,
                                                       (short)0, acc[nt], false, false);
    }
  }
#pragma unroll
  for (int t = 0; t < 2; ++t) {
#pragma unroll
    for (int r = 0; r < 8; ++r) {
      int brow = wv * 16 + r + 8 * hi;
      int jl   = t * 16 + n16;
      int j    = 32 * w + jl;
      float hr = acc[0 + t][r], hz = acc[2 + t][r], hn = acc[4 + t][r];
      float rv = sigf(GIk[j] + hr);
      float zv = sigf(GIk[1024 + j] + hz);
      float nv = tanhf(GIk[2048 + j] + rv * hn);
      float hp = hf_cur[(long)brow * H2 + j];
      float ho = tanhf((1.0f - zv) * nv + zv * hp);
      hf_nxt[(long)brow * H2 + j] = ho;
      int lp = frag_lane(brow, jl), e = frag_elem(jl);
      hdec_nxt[(((long)wv * KT_H2 + w) * 32 + lp) * 16 + e] = (_Float16)ho;
    }
  }
}

// ---------------------------------------------------------------------------
// logits = h_out @ cls_W.T + cls_b, then log_softmax over the 2 classes.
__global__ void k_logits(const float* __restrict__ hf, const float* __restrict__ clsW,
                         const float* __restrict__ clsB, float* __restrict__ out) {
  __shared__ float lg[256];
  int tid = threadIdx.x, row = tid >> 1, c = tid & 1;
  const float* hr = hf + (long)row * H2;
  const float* wr = clsW + (long)c * H2;
  float a = clsB[c];
  for (int j = 0; j < H2; ++j) a += hr[j] * wr[j];
  lg[tid] = a;
  __syncthreads();
  float l0 = lg[row * 2], l1 = lg[row * 2 + 1];
  float m = fmaxf(l0, l1);
  float lse = m + logf(expf(l0 - m) + expf(l1 - m));
  out[tid] = ((c == 0) ? l0 : l1) - lse;
}

// ---------------------------------------------------------------------------
extern "C" void kernel_launch(void* const* d_in, const int* in_sizes, int n_in,
                              void* d_out, int out_size, void* d_ws, size_t ws_size,
                              hipStream_t stream) {
  (void)in_sizes; (void)n_in; (void)out_size; (void)ws_size;
  const int*   seq     = (const int*)d_in[0];
  const int*   classes = (const int*)d_in[1];
  const float* embW    = (const float*)d_in[2];
  const float* ecW     = (const float*)d_in[3];
  const float* fWih    = (const float*)d_in[4];
  const float* fWhh    = (const float*)d_in[5];
  const float* fb      = (const float*)d_in[6];
  const float* bWih    = (const float*)d_in[7];
  /* d_in[8] = b_Whh: unused -- backward scan is one step from h=0 */
  const float* bb      = (const float*)d_in[9];
  const float* dWih    = (const float*)d_in[10];
  const float* dWhh    = (const float*)d_in[11];
  const float* dbih    = (const float*)d_in[12];
  const float* dbhh    = (const float*)d_in[13];
  const float* clsW    = (const float*)d_in[14];
  const float* clsB    = (const float*)d_in[15];
  float* out = (float*)d_out;

  char* p = (char*)d_ws;
  auto alloc = [&](size_t bytes) -> char* {
    char* r = p; p += (bytes + 255) & ~(size_t)255; return r;
  };
  int*      bar   = (int*)alloc(256);
  _Float16* X16   = (_Float16*)alloc((size_t)(NROWS / 16) * KT_E * 512 * 2);      // 21 MB
  _Float16* WfihP = (_Float16*)alloc((size_t)(G4 / 16) * KT_E * 512 * 2);
  _Float16* WbihP = (_Float16*)alloc((size_t)(G4 / 16) * KT_E * 512 * 2);
  _Float16* WfhhP = (_Float16*)alloc((size_t)(G4 / 16) * KT_H * 512 * 2);
  _Float16* WdhhP = (_Float16*)alloc((size_t)(G6 / 16) * KT_H2 * 512 * 2);
  _Float16* XGp   = (_Float16*)alloc((size_t)(NROWS / 16) * NT_G4 * 32 * 8 * 2);  // 134 MB
  float*    XB    = (float*)alloc((size_t)Bn * G4 * 4);
  _Float16* hb0   = (_Float16*)alloc((size_t)8 * KT_H * 512 * 2);
  _Float16* hb1   = (_Float16*)alloc((size_t)8 * KT_H * 512 * 2);
  _Float16* hdec0 = (_Float16*)alloc((size_t)8 * KT_H2 * 512 * 2);
  _Float16* hdec1 = (_Float16*)alloc((size_t)8 * KT_H2 * 512 * 2);
  float*    hf0   = (float*)alloc((size_t)Bn * H2 * 4);
  float*    hf1   = (float*)alloc((size_t)Bn * H2 * 4);
  float*    GI    = (float*)alloc((size_t)NC * G6 * 4);

  auto cdiv = [](long a, long b) -> int { return (int)((a + b - 1) / b); };

  size_t smR = (size_t)8 * KT_H * 512 * 2 + (size_t)128 * 32 * 4;  // 144 KB
  size_t smD = (size_t)6 * KT_H2 * 512 * 2;                        // 192 KB
  hipFuncSetAttribute((const void*)k_lstm_fwd,
                      hipFuncAttributeMaxDynamicSharedMemorySize, (int)smR);
  hipFuncSetAttribute((const void*)k_dec_step,
                      hipFuncAttributeMaxDynamicSharedMemorySize, (int)smD);

  // --- prep (parallel, no dependencies on each other) ---
  k_init<<<cdiv(8 * KT_H * 512 / 2 + 256, 256), 256, 0, stream>>>(
      bar, (unsigned int*)hb0, 8 * KT_H * 512 / 2);
  k_pack_b<<<cdiv((long)G4 * EP, 256), 256, 0, stream>>>(fWih, G4, En, KT_E, WfihP);
  k_pack_b<<<cdiv((long)G4 * EP, 256), 256, 0, stream>>>(bWih, G4, En, KT_E, WbihP);
  k_pack_b<<<cdiv((long)G4 * Hn, 256), 256, 0, stream>>>(fWhh, G4, Hn, KT_H, WfhhP);
  k_pack_b<<<cdiv((long)G6 * H2, 256), 256, 0, stream>>>(dWhh, G6, H2, KT_H2, WdhhP);
  k_embed_pack<<<cdiv((long)NROWS * EP, 256), 256, 0, stream>>>(seq, embW, X16);
  k_gi<<<cdiv(NC * G6, 256), 256, 0, stream>>>(classes, ecW, dWih, dbih, GI);

  // --- big hoisted GEMMs ---
  k_gemm_bias<<<dim3(NROWS / 128, G4 / 128), 256, 0, stream>>>(
      X16, WfihP, fb, nullptr, XGp, KT_E, G4, 1);
  k_gemm_bias<<<dim3(1, G4 / 128), 256, 0, stream>>>(
      X16, WbihP, bb, XB, nullptr, KT_E, G4, 0);

  // --- backward "scan" (one cell) + forward recurrence ---
  k_bwd_cell<<<cdiv(Bn * Hn, 256), 256, 0, stream>>>(XB, hdec0, hf0);
  k_lstm_fwd<<<NWGR, 256, smR, stream>>>(WfhhP, XGp, hb0, hb1, hdec0, hf0, bar);

  // --- 6 GRU decoder steps + per-step log-softmax logits ---
  for (int k = 0; k < NC; ++k) {
    const _Float16* hc  = (k & 1) ? hdec1 : hdec0;
    _Float16*       hnx = (k & 1) ? hdec0 : hdec1;
    const float*    hfc = (k & 1) ? hf1 : hf0;
    float*          hfn = (k & 1) ? hf0 : hf1;
    k_dec_step<<<H2 / 32, 256, smD, stream>>>(WdhhP, GI + (long)k * G6, dbhh,
                                              hc, hfc, hnx, hfn);
    k_logits<<<1, 256, 0, stream>>>(hfn, clsW, clsB, out + (long)k * Bn * 2);
  }
}
